// TttModule_1322849927314
// MI455X (gfx1250) — matrix operand order repaired and checked by reading, not compiled, not run on hardware
//
#include <hip/hip_runtime.h>

#define Bv 4
#define Lv 2048
#define Cv 1024
#define NHv 16
#define HDv 64
#define INTERv 256
#define Kv 16
#define NCv 128
#define EPSv 1e-6f

typedef __attribute__((ext_vector_type(16))) __bf16 v16bf;
typedef __attribute__((ext_vector_type(8)))  __bf16 v8bf;
typedef __attribute__((ext_vector_type(8)))  float  v8f;

#if __has_builtin(__builtin_amdgcn_tanhf)
static __device__ __forceinline__ float fast_tanh(float x) {
  return __builtin_amdgcn_tanhf(x);  // v_tanh_f32 (TRANS, co-executes with WMMA)
}
#else
static __device__ __forceinline__ float fast_tanh(float x) { return tanhf(x); }
#endif

static __device__ __forceinline__ v8f zero8() {
  v8f z = {0.f, 0.f, 0.f, 0.f, 0.f, 0.f, 0.f, 0.f};
  return z;
}

static __device__ __forceinline__ v8f wmma_bf16(v16bf a, v16bf b, v8f c) {
  return __builtin_amdgcn_wmma_f32_16x16x32_bf16(false, a, false, b, (short)0, c,
                                                 false, false);
}

// ---- WMMA operand loaders, wave32 layouts ---------------------------------
// fp32 source -> bf16 regs
// A tile (16xK32): A[m][k] = S[m*ld + kb + k]
static __device__ __forceinline__ v16bf ldA(const float* S, int ld, int kb, int kmax) {
  const int lane = threadIdx.x & 31;
  const int row = lane & 15, grp = lane >> 4;
  v16bf a;
#pragma unroll
  for (int e = 0; e < 16; ++e) {
    const int k = kb + (grp << 4) + e;
    a[e] = (__bf16)((k < kmax) ? S[row * ld + k] : 0.0f);
  }
  return a;
}
// A tile from transposed source: A[m][k] = S[k*ld + m0 + m]
static __device__ __forceinline__ v16bf ldAt(const float* S, int ld, int m0, int kmax) {
  const int lane = threadIdx.x & 31;
  const int row = lane & 15, grp = lane >> 4;
  v16bf a;
#pragma unroll
  for (int e = 0; e < 16; ++e) {
    const int k = (grp << 4) + e;
    a[e] = (__bf16)((k < kmax) ? S[k * ld + m0 + row] : 0.0f);
  }
  return a;
}
// B tile (K32x16): B[k][n] = S[k*ld + nb + n]   (row-major B source, strided)
static __device__ __forceinline__ v16bf ldB(const float* S, int ld, int kb, int nb, int kmax) {
  const int lane = threadIdx.x & 31;
  const int n = lane & 15, grp = lane >> 4;
  v16bf b;
#pragma unroll
  for (int e = 0; e < 16; ++e) {
    const int k = kb + (grp << 4) + e;
    b[e] = (__bf16)((k < kmax) ? S[k * ld + nb + n] : 0.0f);
  }
  return b;
}
// B tile = S^T: B[k][n] = S[(nb+n)*ld + k]   (contiguous rows of S)
static __device__ __forceinline__ v16bf ldBt(const float* S, int ld, int kb, int nb, int kmax) {
  const int lane = threadIdx.x & 31;
  const int n = lane & 15, grp = lane >> 4;
  v16bf b;
#pragma unroll
  for (int e = 0; e < 16; ++e) {
    const int k = kb + (grp << 4) + e;
    b[e] = (__bf16)((k < kmax) ? S[(nb + n) * ld + k] : 0.0f);
  }
  return b;
}
// bf16 source, zero-VALU loaders
static __device__ __forceinline__ v16bf ldA_h(const __bf16* S, int ld, int kb) {
  const int lane = threadIdx.x & 31;
  const int row = lane & 15, grp = lane >> 4;
  const __bf16* p = S + row * ld + kb + 16 * grp;
  const v8bf lo = *(const v8bf*)p;
  const v8bf hi = *(const v8bf*)(p + 8);
  v16bf a;
#pragma unroll
  for (int e = 0; e < 8; ++e) {
    a[e] = lo[e];
    a[8 + e] = hi[e];
  }
  return a;
}
// B[k][n] = S[(nb+n)*ld + kb + k] : one 32B contiguous load per lane
static __device__ __forceinline__ v16bf ldBt_h(const __bf16* S, int ld, int kb, int nb) {
  const int lane = threadIdx.x & 31;
  const int n = lane & 15, grp = lane >> 4;
  return *(const v16bf*)(S + (size_t)(nb + n) * ld + kb + (grp << 4));
}

static __device__ __forceinline__ float gelu_t(float x) {
  const float x3 = x * x * x;
  return 0.5f * x * (1.0f + fast_tanh(0.79788456f * (x + 0.044715f * x3)));
}
static __device__ __forceinline__ float dgelu_t(float x) {
  const float t = fast_tanh(0.79788456f * x * (1.0f + 0.044715f * x * x));
  return 0.5f * x * ((1.0f - t * t) * (0.79788456f + 0.1070322243f * x * x)) +
         0.5f * (1.0f + t);
}
static __device__ __forceinline__ float red16(float v) {
  v += __shfl_xor(v, 1, 32);
  v += __shfl_xor(v, 2, 32);
  v += __shfl_xor(v, 4, 32);
  v += __shfl_xor(v, 8, 32);
  return v;
}

// ---------------------------------------------------------------------------
// Kernel 0: fp32 -> bf16 weight conversion (one-time prep into d_ws)
// ---------------------------------------------------------------------------
__global__ __launch_bounds__(256) void k_wcvt(const float* __restrict__ src,
                                              __bf16* __restrict__ dst, int n) {
  const int i = (blockIdx.x * 256 + threadIdx.x) * 4;
  if (i < n) {
    const float4 v = *(const float4*)(src + i);
    dst[i + 0] = (__bf16)v.x;
    dst[i + 1] = (__bf16)v.y;
    dst[i + 2] = (__bf16)v.z;
    dst[i + 3] = (__bf16)v.w;
  }
}

// ---------------------------------------------------------------------------
// Kernel 1: fused QKV projection, software-pipelined k-loop.
// grid (512, 8), block 256 (8 waves, each owns one 16x16 N-tile x {Q,K,V}).
// ---------------------------------------------------------------------------
__global__ __launch_bounds__(256) void k_qkv(const float* __restrict__ X,
                                             const __bf16* __restrict__ Wqh,
                                             const __bf16* __restrict__ Wkh,
                                             const __bf16* __restrict__ Wvh,
                                             float* __restrict__ XA,
                                             float* __restrict__ XB,
                                             float* __restrict__ XC) {
  __shared__ __bf16 sA[16 * Cv];  // 32KB staged A tile
  const int mt = blockIdx.x;
  const int t = threadIdx.x;
  const float* Xrow = X + (size_t)mt * 16 * Cv;
  for (int i = t; i < 16 * Cv; i += 256) sA[i] = (__bf16)Xrow[i];
  __syncthreads();

  const int wid = t >> 5;
  const int nb = blockIdx.y * 128 + wid * 16;
  v8f aq = zero8(), ak = zero8(), av = zero8();

  // 2-stage software pipeline: fetch tiles for kb+32 before WMMAs at kb.
  v16bf a = ldA_h(sA, Cv, 0);
  v16bf bq = ldBt_h(Wqh, Cv, 0, nb);
  v16bf bk = ldBt_h(Wkh, Cv, 0, nb);
  v16bf bv = ldBt_h(Wvh, Cv, 0, nb);
#pragma unroll 4
  for (int kb = 0; kb < Cv; kb += 32) {
    const v16bf a_c = a, bq_c = bq, bk_c = bk, bv_c = bv;
    if (kb + 32 < Cv) {
      a = ldA_h(sA, Cv, kb + 32);
      bq = ldBt_h(Wqh, Cv, kb + 32, nb);
      bk = ldBt_h(Wkh, Cv, kb + 32, nb);
      bv = ldBt_h(Wvh, Cv, kb + 32, nb);
    }
    aq = wmma_bf16(a_c, bq_c, aq);
    ak = wmma_bf16(a_c, bk_c, ak);
    av = wmma_bf16(a_c, bv_c, av);
  }

  const int lane = t & 31, n = lane & 15, grp = lane >> 4;
  const int col = nb + n, h = col >> 6, hd = col & 63;
#pragma unroll
  for (int r = 0; r < 8; ++r) {
    const int m = mt * 16 + r + 8 * grp;
    const int b = m >> 11, l = m & 2047, nc = l >> 4, kk = l & 15;
    const size_t o = ((((size_t)(b * NHv + h)) * NCv + nc) * Kv + kk) * HDv + hd;
    XA[o] = aq[r];
    XB[o] = ak[r];
    XC[o] = av[r];
  }
}

// ---------------------------------------------------------------------------
// Kernel 2: gate logits -> coeff = LR * 1/(k+1) * sigmoid(x.g + b) / HD
// ---------------------------------------------------------------------------
__global__ __launch_bounds__(256) void k_gate(const float* __restrict__ X,
                                              const float* __restrict__ gw,
                                              const float* __restrict__ gb,
                                              float* __restrict__ coeff) {
  const int row = blockIdx.x;
  const int t = threadIdx.x;
  const int h = t >> 4, sub = t & 15;
  const float* xr = X + (size_t)row * Cv;
  const float* wr = gw + (size_t)h * Cv;
  float s = 0.f;
#pragma unroll 8
  for (int j = 0; j < 64; ++j) {
    const int c = sub + 16 * j;
    s += xr[c] * wr[c];
  }
  s = red16(s);
  if (sub == 0) {
    const float sig = 1.0f / (1.0f + __expf(-(s + gb[h])));
    const int b = row >> 11, l = row & 2047, nc = l >> 4, kk = l & 15;
    coeff[(((size_t)(b * NHv + h)) * NCv + nc) * Kv + kk] =
        sig * (1.0f / (float)(kk + 1)) * (1.0f / (float)HDv);
  }
}

// ---------------------------------------------------------------------------
// Kernel 3: per-(batch,head) sequential TTT scan. 64 blocks x 256 threads.
// W1 kept TRANSPOSED in LDS so both W1-GEMMs read contiguous B tiles.
// Next chunk's inputs double-buffered in registers; +2 chunk prefetched to L2.
// ---------------------------------------------------------------------------
__global__ __launch_bounds__(256) void ttt_scan(
    const float* __restrict__ gXA, const float* __restrict__ gXB,
    const float* __restrict__ gXC, const float* __restrict__ gco,
    const float* __restrict__ gW1, const float* __restrict__ gb1,
    const float* __restrict__ gW2, const float* __restrict__ gb2,
    const float* __restrict__ glnw, const float* __restrict__ glnb,
    float* __restrict__ gXCW) {
  __shared__ float sW1t[INTERv * HDv];  // 64KB fp32 master state, W1^T [256][64]
  __shared__ float sW2[INTERv * HDv];   // 64KB fp32 master state, W2 [256][64]
  __shared__ float sb1[INTERv];
  __shared__ float sb2[HDv];
  __shared__ float slnw[HDv];
  __shared__ float slnb[HDv];
  __shared__ float sco[Kv];
  __shared__ float sXA[Kv * HDv];
  __shared__ float sXB[Kv * HDv];
  __shared__ float sXC[Kv * HDv];
  __shared__ float sZ1[Kv * INTERv];
  __shared__ float sX2[Kv * INTERv];
  __shared__ float sG1[Kv * INTERv];
  __shared__ float sX2b[Kv * INTERv];
  __shared__ float sZ2[Kv * HDv];
  __shared__ float sG2[Kv * HDv];
  __shared__ float sM[Kv * Kv];

  const int bh = blockIdx.x;
  const int bb = bh / NHv;
  const int h = bh % NHv;
  const int t = threadIdx.x;
  const int wid = t >> 5;
  const int lane = t & 31;
  const int ln = lane & 15;
  const int lg = lane >> 4;

  // state init (W1 transposed on the fly; coalesced global reads)
  for (int i = t; i < HDv * INTERv; i += 256) {
    const int hd = i >> 8, it = i & 255;  // gW1 linear index i = hd*INTER + it
    sW1t[it * HDv + hd] = gW1[h * HDv * INTERv + i];
  }
  for (int i = t; i < INTERv * HDv; i += 256) sW2[i] = gW2[h * INTERv * HDv + i];
  sb1[t] = gb1[h * INTERv + t];
  if (t < HDv) {
    sb2[t] = gb2[h * HDv + t];
    slnw[t] = glnw[h * HDv + t];
    slnb[t] = glnb[h * HDv + t];
  }

  const size_t hoff = (size_t)bh * NCv * Kv * HDv;
  const float* pXA = gXA + hoff;
  const float* pXB = gXB + hoff;
  const float* pXC = gXC + hoff;
  const float* pco = gco + (size_t)bh * NCv * Kv;

  // preload chunk 0 into registers (Kv*HDv = 1024 floats = 256 thr x float4)
  float4 rA = *(const float4*)(pXA + t * 4);
  float4 rB = *(const float4*)(pXB + t * 4);
  float4 rC = *(const float4*)(pXC + t * 4);
  float rco = (t < Kv) ? pco[t] : 0.f;
  __syncthreads();

  for (int nc = 0; nc < NCv; ++nc) {
    // 1. commit staged chunk to LDS; kick off next chunk's loads
    *(float4*)(sXA + t * 4) = rA;
    *(float4*)(sXB + t * 4) = rB;
    *(float4*)(sXC + t * 4) = rC;
    if (t < Kv) sco[t] = rco;
    __syncthreads();
    if (nc + 1 < NCv) {
      const size_t off = (size_t)(nc + 1) * (Kv * HDv);
      rA = *(const float4*)(pXA + off + t * 4);
      rB = *(const float4*)(pXB + off + t * 4);
      rC = *(const float4*)(pXC + off + t * 4);
      rco = (t < Kv) ? pco[(nc + 1) * Kv + t] : 0.f;
      if (nc + 2 < NCv) {
        const size_t off2 = (size_t)(nc + 2) * (Kv * HDv);
        __builtin_prefetch(pXA + off2 + t * 4, 0, 1);
        __builtin_prefetch(pXB + off2 + t * 4, 0, 1);
        __builtin_prefetch(pXC + off2 + t * 4, 0, 1);
      }
    }

    // 2. Z1 = XB@W1 + b1 ; X2 = gelu(Z1)   [16x256], 8 waves x 2 tiles
    for (int nt = wid * 2; nt < wid * 2 + 2; ++nt) {
      v8f acc = zero8();
      acc = wmma_bf16(ldA(sXB, HDv, 0, HDv), ldBt(sW1t, HDv, 0, nt * 16, HDv), acc);
      acc = wmma_bf16(ldA(sXB, HDv, 32, HDv), ldBt(sW1t, HDv, 32, nt * 16, HDv), acc);
      const int col = nt * 16 + ln;
#pragma unroll
      for (int r = 0; r < 8; ++r) {
        const int m = r + 8 * lg;
        const float z = acc[r] + sb1[col];
        sZ1[m * INTERv + col] = z;
        sX2[m * INTERv + col] = gelu_t(z);
      }
    }
    __syncthreads();

    // 3. Z2 = X2@W2 + b2   [16x64], waves 0..3
    if (wid < 4) {
      v8f acc = zero8();
      for (int kb = 0; kb < INTERv; kb += 32)
        acc = wmma_bf16(ldA(sX2, INTERv, kb, INTERv), ldB(sW2, HDv, kb, wid * 16, INTERv), acc);
      const int col = wid * 16 + ln;
#pragma unroll
      for (int r = 0; r < 8; ++r) sZ2[(r + 8 * lg) * HDv + col] = acc[r] + sb2[col];
    }
    __syncthreads();

    // 4. layernorm fwd + vjp(LN(Z2) - XA) -> G2
    {
      const int row = t >> 4, sub = t & 15;
      float z[4], xh[4], dxh[4];
      float s = 0.f, sq = 0.f;
#pragma unroll
      for (int j = 0; j < 4; ++j) {
        z[j] = sZ2[row * HDv + sub + 16 * j];
        s += z[j];
        sq += z[j] * z[j];
      }
      s = red16(s);
      sq = red16(sq);
      const float mu = s * (1.f / HDv);
      const float var = sq * (1.f / HDv) - mu * mu;
      const float rstd = rsqrtf(var + EPSv);
      float s1 = 0.f, s2 = 0.f;
#pragma unroll
      for (int j = 0; j < 4; ++j) {
        const int c = sub + 16 * j;
        xh[j] = (z[j] - mu) * rstd;
        const float dy = slnw[c] * xh[j] + slnb[c] - sXA[row * HDv + c];
        dxh[j] = dy * slnw[c];
        s1 += dxh[j];
        s2 += dxh[j] * xh[j];
      }
      s1 = red16(s1) * (1.f / HDv);
      s2 = red16(s2) * (1.f / HDv);
#pragma unroll
      for (int j = 0; j < 4; ++j) {
        const int c = sub + 16 * j;
        sG2[row * HDv + c] = (dxh[j] - s1 - xh[j] * s2) * rstd;
      }
    }
    __syncthreads();

    // 5. G1 = (G2 @ W2^T) * dgelu(Z1)   +   6. NegM1 (wave 0)
    for (int nt = wid * 2; nt < wid * 2 + 2; ++nt) {
      v8f acc = zero8();
      acc = wmma_bf16(ldA(sG2, HDv, 0, HDv), ldBt(sW2, HDv, 0, nt * 16, HDv), acc);
      acc = wmma_bf16(ldA(sG2, HDv, 32, HDv), ldBt(sW2, HDv, 32, nt * 16, HDv), acc);
      const int col = nt * 16 + ln;
#pragma unroll
      for (int r = 0; r < 8; ++r) {
        const int m = r + 8 * lg;
        sG1[m * INTERv + col] = acc[r] * dgelu_t(sZ1[m * INTERv + col]);
      }
    }
    if (wid == 0) {  // Attn1 = tril(XC@XB^T); fold coeff & cumsum into NegM1
      v8f acc = zero8();
      acc = wmma_bf16(ldA(sXC, HDv, 0, HDv), ldBt(sXB, HDv, 0, 0, HDv), acc);
      acc = wmma_bf16(ldA(sXC, HDv, 32, HDv), ldBt(sXB, HDv, 32, 0, HDv), acc);
#pragma unroll
      for (int r = 0; r < 8; ++r) {
        const int m = r + 8 * lg;
        sM[m * Kv + ln] = (ln <= m) ? (-sco[m] * (acc[r] + 1.0f)) : 0.0f;
      }
    }
    __syncthreads();

    // 7. Z1b = XC@W1 + NegM1@G1 + b1 ; X2b = gelu(Z1b)
    for (int nt = wid * 2; nt < wid * 2 + 2; ++nt) {
      v8f acc = zero8();
      acc = wmma_bf16(ldA(sXC, HDv, 0, HDv), ldBt(sW1t, HDv, 0, nt * 16, HDv), acc);
      acc = wmma_bf16(ldA(sXC, HDv, 32, HDv), ldBt(sW1t, HDv, 32, nt * 16, HDv), acc);
      acc = wmma_bf16(ldA(sM, Kv, 0, Kv), ldB(sG1, INTERv, 0, nt * 16, Kv), acc);
      const int col = nt * 16 + ln;
#pragma unroll
      for (int r = 0; r < 8; ++r) {
        const int m = r + 8 * lg;
        sX2b[m * INTERv + col] = gelu_t(acc[r] + sb1[col]);
      }
    }
    __syncthreads();

    // 8. NegM2 from Attn2 = tril(X2b @ X2^T) (wave 0)
    if (wid == 0) {
      v8f acc = zero8();
      for (int kb = 0; kb < INTERv; kb += 32)
        acc = wmma_bf16(ldA(sX2b, INTERv, kb, INTERv), ldBt(sX2, INTERv, kb, 0, INTERv), acc);
#pragma unroll
      for (int r = 0; r < 8; ++r) {
        const int m = r + 8 * lg;
        sM[m * Kv + ln] = (ln <= m) ? (-sco[m] * (acc[r] + 1.0f)) : 0.0f;
      }
    }
    __syncthreads();

    // 9. Z2b = X2b@W2 + NegM2@G2 + b2 -> global (waves 0..3)
    if (wid < 4) {
      v8f acc = zero8();
      for (int kb = 0; kb < INTERv; kb += 32)
        acc = wmma_bf16(ldA(sX2b, INTERv, kb, INTERv), ldB(sW2, HDv, kb, wid * 16, INTERv), acc);
      acc = wmma_bf16(ldA(sM, Kv, 0, Kv), ldB(sG2, HDv, 0, wid * 16, Kv), acc);
      const int col = wid * 16 + ln;
#pragma unroll
      for (int r = 0; r < 8; ++r) {
        const int m = r + 8 * lg;
        gXCW[((size_t)(bb * Lv + nc * Kv + m)) * Cv + h * HDv + col] = acc[r] + sb2[col];
      }
    }
    __syncthreads();

    // 10. state updates: W1 -= c*XB^T@G1 (stored transposed), W2 -= c*X2^T@G2
    {
      const float clast = sco[Kv - 1];
      for (int tile = wid; tile < 64; tile += 8) {  // W1: 4 (M=HD) x 16 (N=INTER)
        const int mt = tile & 3, nt2 = tile >> 2;
        v8f acc = zero8();
        acc = wmma_bf16(ldAt(sXB, HDv, mt * 16, Kv), ldB(sG1, INTERv, 0, nt2 * 16, Kv), acc);
        const int cn = nt2 * 16 + ln;
#pragma unroll
        for (int r = 0; r < 8; ++r)
          sW1t[cn * HDv + (mt * 16 + r + 8 * lg)] -= clast * acc[r];
      }
      for (int tile = wid; tile < 64; tile += 8) {  // W2: 16 (M=INTER) x 4 (N=HD)
        const int mt = tile >> 2, nt2 = tile & 3;
        v8f acc = zero8();
        acc = wmma_bf16(ldAt(sX2, INTERv, mt * 16, Kv), ldB(sG2, HDv, 0, nt2 * 16, Kv), acc);
        const int cn = nt2 * 16 + ln;
#pragma unroll
        for (int r = 0; r < 8; ++r)
          sW2[(mt * 16 + r + 8 * lg) * HDv + cn] -= clast * acc[r];
      }
      float s = 0.f;
#pragma unroll
      for (int k2 = 0; k2 < Kv; ++k2) s += sG1[k2 * INTERv + t];
      sb1[t] -= clast * s;
      if (t < HDv) {
        float s2b = 0.f;
#pragma unroll
        for (int k2 = 0; k2 < Kv; ++k2) s2b += sG2[k2 * HDv + t];
        sb2[t] -= clast * s2b;
      }
    }
    __syncthreads();
  }
}

// ---------------------------------------------------------------------------
// Kernel 4: out = XCW @ Wo^T, software-pipelined. grid (512, 8), block 256.
// ---------------------------------------------------------------------------
__global__ __launch_bounds__(256) void k_out(const float* __restrict__ XCW,
                                             const __bf16* __restrict__ Woh,
                                             float* __restrict__ Y) {
  __shared__ __bf16 sA[16 * Cv];
  const int mt = blockIdx.x;
  const int t = threadIdx.x;
  const float* Xrow = XCW + (size_t)mt * 16 * Cv;
  for (int i = t; i < 16 * Cv; i += 256) sA[i] = (__bf16)Xrow[i];
  __syncthreads();

  const int wid = t >> 5;
  const int nb = blockIdx.y * 128 + wid * 16;
  v8f acc = zero8();
  v16bf a = ldA_h(sA, Cv, 0);
  v16bf bo = ldBt_h(Woh, Cv, 0, nb);
#pragma unroll 4
  for (int kb = 0; kb < Cv; kb += 32) {
    const v16bf a_c = a, bo_c = bo;
    if (kb + 32 < Cv) {
      a = ldA_h(sA, Cv, kb + 32);
      bo = ldBt_h(Woh, Cv, kb + 32, nb);
    }
    acc = wmma_bf16(a_c, bo_c, acc);
  }

  const int lane = t & 31, n = lane & 15, grp = lane >> 4;
#pragma unroll
  for (int r = 0; r < 8; ++r)
    Y[((size_t)(mt * 16 + r + 8 * grp)) * Cv + nb + n] = acc[r];
}

extern "C" void kernel_launch(void* const* d_in, const int* in_sizes, int n_in,
                              void* d_out, int out_size, void* d_ws, size_t ws_size,
                              hipStream_t stream) {
  (void)in_sizes; (void)n_in; (void)out_size; (void)ws_size;
  const float* X   = (const float*)d_in[0];
  const float* Wq  = (const float*)d_in[1];
  const float* Wk  = (const float*)d_in[2];
  const float* Wv  = (const float*)d_in[3];
  const float* Wo  = (const float*)d_in[4];
  const float* W1  = (const float*)d_in[5];
  const float* b1  = (const float*)d_in[6];
  const float* W2  = (const float*)d_in[7];
  const float* b2  = (const float*)d_in[8];
  const float* lnw = (const float*)d_in[9];
  const float* lnb = (const float*)d_in[10];
  const float* gw  = (const float*)d_in[11];
  const float* gb  = (const float*)d_in[12];

  float* wsf = (float*)d_ws;
  const size_t SZ = (size_t)Bv * NHv * NCv * Kv * HDv;  // 8,388,608
  const size_t SZCO = (size_t)Bv * NHv * NCv * Kv;      // 131,072
  const size_t WSZ = (size_t)Cv * Cv;                   // 1,048,576
  float* XA = wsf;
  float* XB = XA + SZ;
  float* XC = XB + SZ;
  float* CO = XC + SZ;
  float* XCW = CO + SZCO;
  __bf16* Wqh = (__bf16*)(XCW + SZ);
  __bf16* Wkh = Wqh + WSZ;
  __bf16* Wvh = Wkh + WSZ;
  __bf16* Woh = Wvh + WSZ;

  const int cvtGrid = (int)(WSZ / (256 * 4));  // 1024
  k_wcvt<<<dim3(cvtGrid), dim3(256), 0, stream>>>(Wq, Wqh, (int)WSZ);
  k_wcvt<<<dim3(cvtGrid), dim3(256), 0, stream>>>(Wk, Wkh, (int)WSZ);
  k_wcvt<<<dim3(cvtGrid), dim3(256), 0, stream>>>(Wv, Wvh, (int)WSZ);
  k_wcvt<<<dim3(cvtGrid), dim3(256), 0, stream>>>(Wo, Woh, (int)WSZ);

  k_qkv<<<dim3(Bv * Lv / 16, Cv / 128), dim3(256), 0, stream>>>(X, Wqh, Wkh, Wvh, XA, XB, XC);
  k_gate<<<dim3(Bv * Lv), dim3(256), 0, stream>>>(X, gw, gb, CO);
  ttt_scan<<<dim3(Bv * NHv), dim3(256), 0, stream>>>(XA, XB, XC, CO, W1, b1, W2, b2,
                                                     lnw, lnb, XCW);
  k_out<<<dim3(Bv * Lv / 16, Cv / 128), dim3(256), 0, stream>>>(XCW, Woh, (float*)d_out);
}
